// DoubleGCN_53712861003780
// MI455X (gfx1250) — matrix operand order repaired and checked by reading, not compiled, not run on hardware
//
#include <hip/hip_runtime.h>
#include <stdint.h>

typedef __attribute__((ext_vector_type(16))) __bf16 v16bf;
typedef __attribute__((ext_vector_type(8)))  __bf16 v8bf;
typedef __attribute__((ext_vector_type(4)))  __bf16 v4bf;
typedef __attribute__((ext_vector_type(8)))  float  v8f;
typedef __attribute__((ext_vector_type(4)))  float  v4f;
typedef __attribute__((ext_vector_type(2)))  float  v2f;

#define FEAT 256
#define HID  128
#define OUTD 64

__global__ void zero_f32(float* __restrict__ p, int n) {
  int i = blockIdx.x * blockDim.x + threadIdx.x;
  if (i < n) p[i] = 0.0f;
}

__global__ void degree_accum(const int* __restrict__ dst, float* __restrict__ deg, int E) {
  int i = blockIdx.x * blockDim.x + threadIdx.x;
  if (i < E) atomicAdd(&deg[dst[i]], 1.0f);
}

__global__ void deg_to_dinv(float* __restrict__ deg, int n) {
  int i = blockIdx.x * blockDim.x + threadIdx.x;
  if (i < n) deg[i] = rsqrtf(deg[i] + 1.0f);   // +1 self loop => deg >= 1 always
}

// Streaming f32 -> bf16 (4 elems/thread, packed converts). n must be multiple of 4 (it is).
__global__ void cvt_f32_bf16(const float* __restrict__ x, __bf16* __restrict__ y, size_t n4) {
  size_t i = (size_t)blockIdx.x * blockDim.x + threadIdx.x;
  if (i >= n4) return;
  v4f v = ((const v4f*)x)[i];
  ((v4bf*)y)[i] = __builtin_convertvector(v, v4bf);
}

// W[K,N] f32 -> Wt[N,K] bf16 (tiny: <= 32K elements)
__global__ void transpose_f32_bf16(const float* __restrict__ W, __bf16* __restrict__ Wt,
                                   int K, int N) {
  int i = blockIdx.x * blockDim.x + threadIdx.x;
  if (i >= K * N) return;
  int k = i / N, c = i % N;
  Wt[(size_t)c * K + k] = (__bf16)W[i];
}

// C[M,N] = A[M,K] * W[K,N]; A row-major bf16, Wt = W^T [N,K] bf16, C f32.
// One wave per 16x16 C tile; K compile-time so the loop fully unrolls:
// per K-step = 4x global_load_b128 + 1x v_wmma_f32_16x16x32_bf16.
template<int K, int N>
__global__ void gemm_bf16_wmma(const __bf16* __restrict__ A, const __bf16* __restrict__ Wt,
                               float* __restrict__ C, int M) {
  const int lane = threadIdx.x & 31;
  const int half = lane >> 4;     // ISA 16-bit A/B fragment: lanes 16-31 hold the +8/+16 K half
  const int l15  = lane & 15;
  const int mBase = blockIdx.x << 4;
  const int nBase = blockIdx.y << 4;

  v8f acc = {};
  const __bf16* arow = A  + (size_t)(mBase + l15) * K + half * 8;
  const __bf16* brow = Wt + (size_t)(nBase + l15) * K + half * 8;

  #pragma unroll
  for (int k0 = 0; k0 < K; k0 += 32) {
    // A 16x32: elems 0..7 <-> K=k0+half*8+0..7 ; elems 8..15 <-> +16
    v8bf alo = *(const v8bf*)(arow + k0);
    v8bf ahi = *(const v8bf*)(arow + k0 + 16);
    // B 32x16: elems 0..7 <-> K=k0+half*8+0..7 ; elems 8..15 <-> +16, N=nBase+l15
    v8bf blo = *(const v8bf*)(brow + k0);
    v8bf bhi = *(const v8bf*)(brow + k0 + 16);
    v16bf af = __builtin_shufflevector(alo, ahi, 0,1,2,3,4,5,6,7,8,9,10,11,12,13,14,15);
    v16bf bf = __builtin_shufflevector(blo, bhi, 0,1,2,3,4,5,6,7,8,9,10,11,12,13,14,15);
    if (k0 + 32 < K) {                          // global_prefetch_b8 of next K-slab
      __builtin_prefetch(arow + k0 + 32, 0, 3);
      __builtin_prefetch(brow + k0 + 32, 0, 3);
    }
    acc = __builtin_amdgcn_wmma_f32_16x16x32_bf16(
        false, af, false, bf, (short)0, acc, false, false);
  }

  // C/D layout: lanes 0-15 rows M=r, lanes 16-31 rows M=8+r; N = l15
  float* crow = C + (size_t)(mBase + half * 8) * N + (nBase + l15);
  #pragma unroll
  for (int r = 0; r < 8; ++r) crow[(size_t)r * N] = acc[r];
}

// agg[i,c] = bias[c] + h[i,c] * dinv[i]^2   (self-loop term; full overwrite of agg)
__global__ void init_agg(const float* __restrict__ h, const float* __restrict__ dinv,
                         const float* __restrict__ bias, float* __restrict__ agg,
                         int n, int C) {
  size_t i = (size_t)blockIdx.x * blockDim.x + threadIdx.x;
  if (i >= (size_t)n * C) return;
  int row = (int)(i / C), c = (int)(i % C);
  float di = dinv[row];
  agg[i] = bias[c] + h[i] * di * di;
}

// One wave per edge: agg[dst] += h[src] * dinv[src] * dinv[dst]
// Vector load of the per-lane slice, then float atomics (accumulator lives in L2).
template<int C>
__global__ void edge_scatter(const int* __restrict__ src, const int* __restrict__ dst,
                             const float* __restrict__ h, const float* __restrict__ dinv,
                             float* __restrict__ agg, int E) {
  constexpr int per = C >> 5;          // 4 (HID) or 2 (OUTD)
  int wpb = blockDim.x >> 5;
  int e = blockIdx.x * wpb + (threadIdx.x >> 5);
  if (e >= E) return;
  int lane = threadIdx.x & 31;
  int s = src[e], d = dst[e];
  float norm = dinv[s] * dinv[d];
  const float* hs = h + (size_t)s * C + lane * per;
  float* ad = agg + (size_t)d * C + lane * per;
  if constexpr (per == 4) {
    v4f v = *(const v4f*)hs;
    atomicAdd(&ad[0], v[0] * norm);
    atomicAdd(&ad[1], v[1] * norm);
    atomicAdd(&ad[2], v[2] * norm);
    atomicAdd(&ad[3], v[3] * norm);
  } else {
    v2f v = *(const v2f*)hs;
    atomicAdd(&ad[0], v[0] * norm);
    atomicAdd(&ad[1], v[1] * norm);
  }
}

// Per-feature sum & sumsq (register partials, few atomics). sums[0..C)=sum, [C..2C)=sumsq
__global__ void bn_stats(const float* __restrict__ x, float* __restrict__ sums, int n, int C) {
  int tpf = blockDim.x / C;
  int c   = threadIdx.x % C;
  int sub = threadIdx.x / C;
  float s = 0.0f, s2 = 0.0f;
  for (int row = blockIdx.x * tpf + sub; row < n; row += gridDim.x * tpf) {
    float v = x[(size_t)row * C + c];
    s += v; s2 += v * v;
  }
  atomicAdd(&sums[c], s);
  atomicAdd(&sums[C + c], s2);
}

__global__ void bn_relu(float* __restrict__ x, const float* __restrict__ sums,
                        const float* __restrict__ gamma, const float* __restrict__ beta,
                        int n, int C) {
  size_t i = (size_t)blockIdx.x * blockDim.x + threadIdx.x;
  if (i >= (size_t)n * C) return;
  int c = (int)(i % C);
  float inv_n = 1.0f / (float)n;
  float mu  = sums[c] * inv_n;
  float var = sums[C + c] * inv_n - mu * mu;       // biased variance
  float v = (x[i] - mu) * rsqrtf(var + 1e-5f) * gamma[c] + beta[c];
  x[i] = v > 0.0f ? v : 0.0f;
}

// One wave per batch node, C == 64 (2 elems/lane), wave32 shuffle reductions.
__global__ void gather_logsoftmax(const float* __restrict__ x, const int* __restrict__ nodes,
                                  float* __restrict__ out, int B) {
  int wpb = blockDim.x >> 5;
  int b = blockIdx.x * wpb + (threadIdx.x >> 5);
  if (b >= B) return;
  int lane = threadIdx.x & 31;
  const float* row = x + (size_t)nodes[b] * OUTD;
  float v0 = row[lane], v1 = row[lane + 32];
  float m = fmaxf(v0, v1);
  #pragma unroll
  for (int off = 16; off > 0; off >>= 1) m = fmaxf(m, __shfl_xor(m, off, 32));
  float s = __expf(v0 - m) + __expf(v1 - m);
  #pragma unroll
  for (int off = 16; off > 0; off >>= 1) s += __shfl_xor(s, off, 32);
  float lse = m + __logf(s);
  out[(size_t)b * OUTD + lane]      = v0 - lse;
  out[(size_t)b * OUTD + lane + 32] = v1 - lse;
}

extern "C" void kernel_launch(void* const* d_in, const int* in_sizes, int n_in,
                              void* d_out, int out_size, void* d_ws, size_t ws_size,
                              hipStream_t stream) {
  (void)n_in; (void)out_size; (void)ws_size;
  const float* features = (const float*)d_in[0];
  const int*   edge     = (const int*)d_in[1];
  const int*   batch    = (const int*)d_in[2];
  const float* W1  = (const float*)d_in[3];
  const float* b1  = (const float*)d_in[4];
  const float* g1  = (const float*)d_in[5];
  const float* be1 = (const float*)d_in[6];
  const float* W2  = (const float*)d_in[7];
  const float* b2  = (const float*)d_in[8];
  const float* g2  = (const float*)d_in[9];
  const float* be2 = (const float*)d_in[10];
  float* out = (float*)d_out;

  const int n = in_sizes[0] / FEAT;      // 100000
  const int E = in_sizes[1] / 2;         // 1600000
  const int B = in_sizes[2];             // 10000
  const int* src = edge;
  const int* dst = edge + E;

  // workspace layout (fully rewritten every call)
  float*  bufA  = (float*)d_ws;                      // n*HID f32  (h1, then x2)
  float*  bufB  = bufA + (size_t)n * HID;            // n*HID f32  (agg1/a1, then agg2/a2)
  float*  dinv  = bufB + (size_t)n * HID;            // n f32
  float*  stats = dinv + n;                          // 2*HID f32
  __bf16* Xbf   = (__bf16*)(stats + 2 * HID);        // n*FEAT bf16 (GEMM A operand)
  __bf16* Wt    = (__bf16*)(Xbf + (size_t)n * FEAT); // FEAT*HID bf16 (transposed weight)

  const size_t nH = (size_t)n * HID, nO = (size_t)n * OUTD;
  const size_t nF4 = (size_t)n * FEAT / 4, nH4 = nH / 4;

  // degree / normalization
  zero_f32<<<(n + 255) / 256, 256, 0, stream>>>(dinv, n);
  degree_accum<<<(E + 255) / 256, 256, 0, stream>>>(dst, dinv, E);
  deg_to_dinv<<<(n + 255) / 256, 256, 0, stream>>>(dinv, n);

  // layer 1: h1 = X @ W1 (bf16 WMMA), agg = b1 + self-loop + edge scatter, BN+ReLU
  cvt_f32_bf16<<<(int)((nF4 + 255) / 256), 256, 0, stream>>>(features, Xbf, nF4);
  transpose_f32_bf16<<<(FEAT * HID + 255) / 256, 256, 0, stream>>>(W1, Wt, FEAT, HID);
  gemm_bf16_wmma<FEAT, HID><<<dim3(n / 16, HID / 16), 32, 0, stream>>>(Xbf, Wt, bufA, n);
  init_agg<<<(int)((nH + 255) / 256), 256, 0, stream>>>(bufA, dinv, b1, bufB, n, HID);
  edge_scatter<HID><<<(E + 7) / 8, 256, 0, stream>>>(src, dst, bufA, dinv, bufB, E);
  zero_f32<<<1, 256, 0, stream>>>(stats, 2 * HID);
  bn_stats<<<512, 256, 0, stream>>>(bufB, stats, n, HID);
  bn_relu<<<(int)((nH + 255) / 256), 256, 0, stream>>>(bufB, stats, g1, be1, n, HID);

  // layer 2: x2 = a1 @ W2 (bf16 WMMA), agg2, BN+ReLU
  cvt_f32_bf16<<<(int)((nH4 + 255) / 256), 256, 0, stream>>>(bufB, Xbf, nH4);
  transpose_f32_bf16<<<(HID * OUTD + 255) / 256, 256, 0, stream>>>(W2, Wt, HID, OUTD);
  gemm_bf16_wmma<HID, OUTD><<<dim3(n / 16, OUTD / 16), 32, 0, stream>>>(Xbf, Wt, bufA, n);
  init_agg<<<(int)((nO + 255) / 256), 256, 0, stream>>>(bufA, dinv, b2, bufB, n, OUTD);
  edge_scatter<OUTD><<<(E + 7) / 8, 256, 0, stream>>>(src, dst, bufA, dinv, bufB, E);
  zero_f32<<<1, 256, 0, stream>>>(stats, 2 * OUTD);
  bn_stats<<<512, 256, 0, stream>>>(bufB, stats, n, OUTD);
  bn_relu<<<(int)((nO + 255) / 256), 256, 0, stream>>>(bufB, stats, g2, be2, n, OUTD);

  // batch gather + log_softmax
  gather_logsoftmax<<<(B + 7) / 8, 256, 0, stream>>>(bufB, batch, out, B);
}
